// DAEEmbeddingModel_46248207843741
// MI455X (gfx1250) — matrix operand order repaired and checked by reading, not compiled, not run on hardware
//
#include <hip/hip_runtime.h>
#include <hip/hip_bf16.h>

// ---------------------------------------------------------------------------
// DAE embedding model forward for MI455X (gfx1250, wave32, WMMA).
//   y  = sigmoid( gather-sum(W_enc, x_cat) + b_enc )        [2048 x 128]  fp32
//   z  = y @ W_dec^T + b_dec                                [2048 x 64000] fp32
// GEMM runs in f16 inputs / f32 accumulate via v_wmma_f32_16x16x32_f16.
// z stores are non-temporal: the 524 MB output stream must not evict the
// L2-resident f16 W_dec (16 MB, reused by all 128 M-tiles).
// ---------------------------------------------------------------------------

#define BATCH      2048
#define NCOLS      32
#define CATDIM     2000
#define HIDDEN     128
#define TOTAL      64000          // NCOLS * CATDIM
#define NTILE      64             // N columns per wave (4 x 16)
#define MTILES     (BATCH / 16)   // 128
#define NGROUPS    (TOTAL / NTILE)// 1000

typedef __attribute__((ext_vector_type(16))) _Float16 v16h;
typedef __attribute__((ext_vector_type(8)))  _Float16 v8h;
typedef __attribute__((ext_vector_type(4)))  _Float16 v4h;
typedef __attribute__((ext_vector_type(8)))  float    v8f;
typedef __attribute__((ext_vector_type(4)))  float    v4f;

// Build a v16h WMMA operand from two contiguous 16-byte chunks.
__device__ __forceinline__ v16h load_op16(const _Float16* p0,
                                          const _Float16* p1) {
  const v8h a = *(const v8h*)p0;
  const v8h b = *(const v8h*)p1;
  v16h r;
#pragma unroll
  for (int j = 0; j < 8; ++j) { r[j] = a[j]; r[8 + j] = b[j]; }
  return r;
}

// ---------------------------------------------------------------------------
// Kernel 0: convert W_dec (fp32 [64000 x 128], row-major) to f16 in d_ws.
// One-shot streaming read -> non-temporal loads.
// ---------------------------------------------------------------------------
__global__ void dae_cast_wdec_f16(const float* __restrict__ wdec,
                                  _Float16* __restrict__ w16) {
  const size_t i4 = (size_t)blockIdx.x * blockDim.x + threadIdx.x; // float4 idx
  const v4f* src = (const v4f*)(wdec + i4 * 4);
  const v4f v = __builtin_nontemporal_load(src);
  v4h h;
#pragma unroll
  for (int j = 0; j < 4; ++j) h[j] = (_Float16)v[j];
  *(v4h*)(w16 + i4 * 4) = h;
}

// ---------------------------------------------------------------------------
// Kernel 1: encoder. One block (128 threads) per batch row; thread = hidden h.
// y_pre[b,h] = b_enc[h] + sum_i W_enc[h, i*2000 + x_cat[b,i]];  y = sigmoid.
// Writes fp32 y to d_out and f16 copy to workspace for the WMMA GEMM.
// ---------------------------------------------------------------------------
__global__ void dae_encode(const int*   __restrict__ x_cat,
                           const float* __restrict__ W_enc,   // [128 x 64000]
                           const float* __restrict__ b_enc,   // [128]
                           float*       __restrict__ y_out,   // [2048 x 128]
                           _Float16*    __restrict__ y16) {   // [2048 x 128]
  const int b = blockIdx.x;
  const int h = threadIdx.x;
  const int* xrow = x_cat + b * NCOLS;
  const float* wrow = W_enc + (size_t)h * TOTAL;

  float sum = b_enc[h];
#pragma unroll
  for (int i = 0; i < NCOLS; ++i) {
    const int g = xrow[i] + i * CATDIM;   // global one-hot index for column i
    sum += wrow[g];
  }
  const float yv = 1.0f / (1.0f + __expf(-sum));
  y_out[b * HIDDEN + h] = yv;
  y16[b * HIDDEN + h]   = (_Float16)yv;
}

// ---------------------------------------------------------------------------
// Kernel 2: decoder GEMM  z[b,v] = sum_h y16[b,h] * w16[v,h] + b_dec[v].
// Wave32: each wave computes a 16(M) x 64(N) tile with 4 v8f accumulators.
// K = 128 in 4 chunks of 32, double-buffered: while the 4 WMMAs of chunk kk
// execute, the A/B b128 loads for chunk kk+1 are already in flight.
// ---------------------------------------------------------------------------
__global__ __launch_bounds__(256)
void dae_decode_wmma(const _Float16* __restrict__ y16,   // [2048 x 128]
                     const _Float16* __restrict__ w16,   // [64000 x 128]
                     const float*    __restrict__ b_dec, // [64000]
                     float*          __restrict__ z) {   // [2048 x 64000]
  const int wave_in_blk = threadIdx.x >> 5;
  const int lane        = threadIdx.x & 31;
  const int hi          = lane >> 4;   // 0: lanes 0-15, 1: lanes 16-31
  const int l16         = lane & 15;

  const int wave  = blockIdx.x * 8 + wave_in_blk;   // 0 .. 127999
  const int mTile = wave % MTILES;                  // fastest -> L2 reuse of B
  const int nGrp  = wave / MTILES;
  const int m0    = mTile * 16;
  const int n0    = nGrp * NTILE;

  // Per-lane base pointers.
  // A-matrix 16x32 f16 layout: lane holds K = 8*hi+0..7 and 16+8*hi+0..7.
  const _Float16* arow = y16 + (size_t)(m0 + l16) * HIDDEN + 8 * hi;
  // B-matrix 32x16 f16 layout: lane holds K = 8*hi+0..7 and 16+8*hi+0..7 at N = l16.
  const _Float16* brow[4];
#pragma unroll
  for (int t = 0; t < 4; ++t)
    brow[t] = w16 + (size_t)(n0 + 16 * t + l16) * HIDDEN + 8 * hi;

  // Accumulators initialized with the bias (bias depends only on column).
  v8f acc[4];
#pragma unroll
  for (int t = 0; t < 4; ++t) {
    const float bv = b_dec[n0 + 16 * t + l16];
    acc[t] = (v8f){bv, bv, bv, bv, bv, bv, bv, bv};
  }

  // Prologue: load K-chunk 0 operands.
  v16h aCur = load_op16(arow, arow + 16);
  v16h bCur[4];
#pragma unroll
  for (int t = 0; t < 4; ++t)
    bCur[t] = load_op16(brow[t], brow[t] + 16);

#pragma unroll
  for (int kk = 0; kk < 4; ++kk) {
    v16h aNxt, bNxt[4];
    if (kk < 3) {
      const int k0 = (kk + 1) * 32;
      aNxt = load_op16(arow + k0, arow + k0 + 16);
#pragma unroll
      for (int t = 0; t < 4; ++t)
        bNxt[t] = load_op16(brow[t] + k0, brow[t] + k0 + 16);
    }

#pragma unroll
    for (int t = 0; t < 4; ++t) {
      acc[t] = __builtin_amdgcn_wmma_f32_16x16x32_f16(
          /*neg_a=*/false, aCur, /*neg_b=*/false, bCur[t],
          /*c_mod=*/(short)0, acc[t], /*reuse_a=*/false, /*reuse_b=*/false);
    }

    if (kk < 3) {
      aCur = aNxt;
#pragma unroll
      for (int t = 0; t < 4; ++t) bCur[t] = bNxt[t];
    }
  }

  // D layout: VGPR r, lanes 0-15 -> M = r, lanes 16-31 -> M = r + 8; N = l16.
  // Non-temporal: the 524 MB z stream must not evict W_dec from L2.
#pragma unroll
  for (int t = 0; t < 4; ++t) {
    float* zc = z + (size_t)(m0 + 8 * hi) * TOTAL + n0 + 16 * t + l16;
#pragma unroll
    for (int r = 0; r < 8; ++r) {
      __builtin_nontemporal_store(acc[t][r], zc + (size_t)r * TOTAL);
    }
  }
}

// ---------------------------------------------------------------------------
// Launcher. Inputs (setup_inputs order): x_cat, W_enc, b_enc, W_dec, b_dec.
// Output: y (2048*128 fp32) ++ z (2048*64000 fp32).
// Workspace: w16 (64000*128 f16 = 16.0 MiB) then y16 (2048*128 f16 = 512 KiB).
// ---------------------------------------------------------------------------
extern "C" void kernel_launch(void* const* d_in, const int* in_sizes, int n_in,
                              void* d_out, int out_size, void* d_ws, size_t ws_size,
                              hipStream_t stream) {
  const int*   x_cat = (const int*)d_in[0];
  const float* W_enc = (const float*)d_in[1];
  const float* b_enc = (const float*)d_in[2];
  const float* W_dec = (const float*)d_in[3];
  const float* b_dec = (const float*)d_in[4];

  float* y_out = (float*)d_out;
  float* z_out = (float*)d_out + (size_t)BATCH * HIDDEN;

  _Float16* w16 = (_Float16*)d_ws;
  _Float16* y16 = w16 + (size_t)TOTAL * HIDDEN;

  // 0) W_dec -> f16 (8,192,000 elems, 4/thread, 256 threads/block)
  dae_cast_wdec_f16<<<(TOTAL * HIDDEN) / (4 * 256), 256, 0, stream>>>(W_dec, w16);

  // 1) encoder (gather-sum + sigmoid), also emits y16
  dae_encode<<<BATCH, HIDDEN, 0, stream>>>(x_cat, W_enc, b_enc, y_out, y16);

  // 2) decoder GEMM via WMMA: 128 mTiles x 1000 nGroups = 128000 waves,
  //    8 waves per 256-thread block -> 16000 blocks.
  dae_decode_wmma<<<(MTILES * NGROUPS) / 8, 256, 0, stream>>>(y16, w16, b_dec, z_out);
}
